// _System_general_3461743640832
// MI455X (gfx1250) — compile-verified, matrix operand order checked by reading, not run on hardware
//
#include <hip/hip_runtime.h>
#include <stdint.h>

#define NB 16384
#define NX 512
#define NU 256
#define NY 128
#define NQ 256

typedef __attribute__((ext_vector_type(16))) __bf16 v16bf;
typedef __attribute__((ext_vector_type(8)))  __bf16 v8bf;
typedef __attribute__((ext_vector_type(8)))  float  v8f;

// ---------- helpers ----------
static __device__ inline v16bf cat16(v8bf lo, v8bf hi) {
  return __builtin_shufflevector(lo, hi, 0,1,2,3,4,5,6,7,8,9,10,11,12,13,14,15);
}
// load one 16x32 bf16 operand tile row-chunk pair for this lane
static __device__ inline v16bf ld_tile(const __bf16* __restrict__ p) {
  return cat16(*(const v8bf*)p, *(const v8bf*)(p + 16));
}

// ---------- setup kernels ----------
__global__ void k_aug_init(const float* __restrict__ E, float* __restrict__ aug) {
  int i = blockIdx.x * 256 + threadIdx.x;       // 512*1024 elements
  int r = i >> 10, c = i & 1023;
  aug[i] = (c < 512) ? E[(size_t)r * 512 + c] : ((c - 512) == r ? 1.0f : 0.0f);
}

// single-workgroup Gauss-Jordan: aug = [E | I] -> [junk | E^-1]
__global__ __launch_bounds__(1024) void k_gauss(float* __restrict__ aug) {
  __shared__ float s_f[512];
  const int tid = threadIdx.x;
  for (int p = 0; p < 512; ++p) {
    float pv = aug[(size_t)p * 1024 + p];
    __syncthreads();
    float rp = 1.0f / pv;                       // E = I + 0.05N: no pivoting needed
    aug[(size_t)p * 1024 + tid] *= rp;
    __syncthreads();
    if (tid < 512) s_f[tid] = aug[(size_t)tid * 1024 + p];
    float pcol = aug[(size_t)p * 1024 + tid];
    __syncthreads();
    for (int r = 0; r < 512; ++r)
      if (r != p) aug[(size_t)r * 1024 + tid] -= s_f[r] * pcol;
    __syncthreads();
  }
}

// G1=Einv@F (512x512), G2=Einv@B1 (512x256), G3=Einv@B2 (512x256), stored bf16
__global__ void k_precompute_G(const float* __restrict__ aug,
                               const float* __restrict__ F, const float* __restrict__ B1,
                               const float* __restrict__ B2,
                               __bf16* __restrict__ G1, __bf16* __restrict__ G2,
                               __bf16* __restrict__ G3) {
  int idx = blockIdx.x * 256 + threadIdx.x;     // 512*1024 threads
  int n = idx >> 10, c = idx & 1023;
  const float* einv = aug + (size_t)n * 1024 + 512;
  float acc = 0.0f;
  if (c < 512) {
    for (int m = 0; m < 512; ++m) acc += einv[m] * F[(size_t)m * 512 + c];
    G1[(size_t)n * 512 + c] = (__bf16)acc;
  } else if (c < 768) {
    int cc = c - 512;
    for (int m = 0; m < 512; ++m) acc += einv[m] * B1[(size_t)m * 256 + cc];
    G2[(size_t)n * 256 + cc] = (__bf16)acc;
  } else {
    int cc = c - 768;
    for (int m = 0; m < 512; ++m) acc += einv[m] * B2[(size_t)m * 256 + cc];
    G3[(size_t)n * 256 + cc] = (__bf16)acc;
  }
}

__global__ void k_cvt_bf16(const float* __restrict__ s, __bf16* __restrict__ d, int n) {
  int i = blockIdx.x * 256 + threadIdx.x;
  if (i < n) d[i] = (__bf16)s[i];
}

// D11T[k][q] = D11[q][k]  (so the solver reads contiguous rows per step)
__global__ void k_transpose256(const float* __restrict__ s, float* __restrict__ d) {
  int i = blockIdx.x * 256 + threadIdx.x;       // 256*256
  int r = i >> 8, c = i & 255;
  d[(size_t)c * 256 + r] = s[i];
}

// ---------- WMMA GEMM: out[NB,N] = sum_seg A_seg[NB,K] @ W_seg[N,K]^T (+bias) ----------
// wave tile: 64 (M) x 32 (N); 8 f32 accumulators; each A tile feeds 2 B tiles,
// each B tile feeds 4 A tiles. All operands bf16, loaded as b128 pairs.
static __device__ inline void gemm_seg(const __bf16* __restrict__ A,
                                       const __bf16* __restrict__ W, int K,
                                       int m0, int n0, int l16, int koff, v8f* c) {
  for (int k0 = 0; k0 < K; k0 += 32) {
    const __bf16* wp0 = W + (size_t)(n0 + l16) * K + k0 + koff;
    const __bf16* wp1 = W + (size_t)(n0 + 16 + l16) * K + k0 + koff;
    v16bf b0 = ld_tile(wp0);
    v16bf b1 = ld_tile(wp1);
#pragma unroll
    for (int mi = 0; mi < 4; ++mi) {
      const __bf16* ap = A + (size_t)(m0 + mi * 16 + l16) * K + k0 + koff;
      v16bf a = ld_tile(ap);
      c[mi]     = __builtin_amdgcn_wmma_f32_16x16x32_bf16(
          false, a, false, b0, (short)0, c[mi],     false, false);
      c[4 + mi] = __builtin_amdgcn_wmma_f32_16x16x32_bf16(
          false, a, false, b1, (short)0, c[4 + mi], false, false);
    }
  }
}

__global__ __launch_bounds__(256) void k_gemm(
    const __bf16* A0, const __bf16* W0, int K0,
    const __bf16* A1, const __bf16* W1, int K1,
    const __bf16* A2, const __bf16* W2, int K2,
    const float* bias, float* __restrict__ out, int N) {
  int lane = threadIdx.x & 31;
  int wv = threadIdx.x >> 5;
  int gw = blockIdx.x * 8 + wv;                 // one wave = one 64x32 tile
  int m0 = (gw & 255) * 64;                     // 256 m-tiles (NB/64)
  int n0 = (gw >> 8) * 32;
  int l16 = lane & 15;
  int half = lane >> 4;
  int koff = half * 8;

  v8f c[8] = {};
  gemm_seg(A0, W0, K0, m0, n0, l16, koff, c);
  if (A1) gemm_seg(A1, W1, K1, m0, n0, l16, koff, c);
  if (A2) gemm_seg(A2, W2, K2, m0, n0, l16, koff, c);

  float b0 = bias ? bias[n0 + l16] : 0.0f;
  float b1 = bias ? bias[n0 + 16 + l16] : 0.0f;
#pragma unroll
  for (int mi = 0; mi < 4; ++mi)
#pragma unroll
    for (int v = 0; v < 8; ++v) {
      size_t rowoff = (size_t)(m0 + mi * 16 + half * 8 + v) * N;
      out[rowoff + n0 + l16]      = c[mi][v]     + b0;
      out[rowoff + n0 + 16 + l16] = c[4 + mi][v] + b1;
    }
}

// ---------- sequential tanh recurrence (wave-autonomous rank-1 updates) ----------
// wave = 4 batch rows; 8 lanes/row; each lane owns q in [t*32, t*32+32) in VGPRs.
// Emits w directly in bf16 (what the downstream WMMA GEMMs consume).
__global__ __launch_bounds__(256) void k_solver(const float* __restrict__ xu,
                                                const float* __restrict__ D11T,
                                                const float* __restrict__ lam,
                                                __bf16* __restrict__ wh) {
  int lane = threadIdx.x & 31;
  int wv = threadIdx.x >> 5;
  int rgrp = lane >> 3;
  int t = lane & 7;
  int row = blockIdx.x * 32 + wv * 4 + rgrp;
  int base = lane & ~7;

  float acc[32];
  const float4* x4 = (const float4*)(xu + (size_t)row * NQ + t * 32);
#pragma unroll
  for (int i = 0; i < 8; ++i) {
    float4 v = x4[i];
    acc[4 * i + 0] = v.x; acc[4 * i + 1] = v.y;
    acc[4 * i + 2] = v.z; acc[4 * i + 3] = v.w;
  }

  for (int to = 0; to < 8; ++to) {
#pragma unroll
    for (int j = 0; j < 32; ++j) {
      int k = to * 32 + j;
      // broadcast acc[j] from the owner lane (t == to) of this row group
      float raw = __int_as_float(
          __builtin_amdgcn_ds_bpermute((base + to) << 2, __float_as_int(acc[j])));
      float wk = tanhf(raw / lam[k]);
      if (t == to) wh[(size_t)row * NQ + k] = (__bf16)wk;
      const float4* d4 = (const float4*)(D11T + (size_t)k * NQ + t * 32);
#pragma unroll
      for (int i = 0; i < 8; ++i) {
        float4 dv = d4[i];
        acc[4 * i + 0] = fmaf(wk, dv.x, acc[4 * i + 0]);
        acc[4 * i + 1] = fmaf(wk, dv.y, acc[4 * i + 1]);
        acc[4 * i + 2] = fmaf(wk, dv.z, acc[4 * i + 2]);
        acc[4 * i + 3] = fmaf(wk, dv.w, acc[4 * i + 3]);
      }
    }
  }
}

// ---------- launch ----------
extern "C" void kernel_launch(void* const* d_in, const int* in_sizes, int n_in,
                              void* d_out, int out_size, void* d_ws, size_t ws_size,
                              hipStream_t stream) {
  (void)in_sizes; (void)n_in; (void)out_size; (void)ws_size;
  const float* x   = (const float*)d_in[0];
  const float* u   = (const float*)d_in[1];
  const float* F   = (const float*)d_in[2];
  const float* B1  = (const float*)d_in[3];
  const float* B2  = (const float*)d_in[4];
  const float* C1  = (const float*)d_in[5];
  const float* C2  = (const float*)d_in[6];
  const float* D11 = (const float*)d_in[7];
  const float* D12 = (const float*)d_in[8];
  const float* D21 = (const float*)d_in[9];
  const float* E   = (const float*)d_in[10];
  const float* lam = (const float*)d_in[11];
  const float* bv  = (const float*)d_in[12];

  char* ws = (char*)d_ws;
  size_t off = 0;
  auto alloc = [&](size_t bytes) -> char* {
    char* p = ws + off;
    off = (off + bytes + 255) & ~(size_t)255;
    return p;
  };
  float*  aug  = (float*) alloc((size_t)512 * 1024 * 4);
  float*  xu   = (float*) alloc((size_t)NB * NQ * 4);
  float*  D11T = (float*) alloc((size_t)256 * 256 * 4);
  __bf16* xh   = (__bf16*)alloc((size_t)NB * NX * 2);
  __bf16* uh   = (__bf16*)alloc((size_t)NB * NU * 2);
  __bf16* wh   = (__bf16*)alloc((size_t)NB * NQ * 2);
  __bf16* G1h  = (__bf16*)alloc((size_t)512 * 512 * 2);
  __bf16* G2h  = (__bf16*)alloc((size_t)512 * 256 * 2);
  __bf16* G3h  = (__bf16*)alloc((size_t)512 * 256 * 2);
  __bf16* C1h  = (__bf16*)alloc((size_t)256 * 512 * 2);
  __bf16* D12h = (__bf16*)alloc((size_t)256 * 256 * 2);
  __bf16* C2h  = (__bf16*)alloc((size_t)128 * 512 * 2);
  __bf16* D21h = (__bf16*)alloc((size_t)128 * 256 * 2);

  float* dx = (float*)d_out;
  float* y  = (float*)d_out + (size_t)NB * NX;

  // small precomputes
  k_aug_init<<<512 * 1024 / 256, 256, 0, stream>>>(E, aug);
  k_gauss<<<1, 1024, 0, stream>>>(aug);
  k_precompute_G<<<512 * 1024 / 256, 256, 0, stream>>>(aug, F, B1, B2, G1h, G2h, G3h);
  k_cvt_bf16<<<(256 * 512) / 256, 256, 0, stream>>>(C1, C1h, 256 * 512);
  k_cvt_bf16<<<(256 * 256) / 256, 256, 0, stream>>>(D12, D12h, 256 * 256);
  k_cvt_bf16<<<(128 * 512) / 256, 256, 0, stream>>>(C2, C2h, 128 * 512);
  k_cvt_bf16<<<(128 * 256) / 256, 256, 0, stream>>>(D21, D21h, 128 * 256);
  k_transpose256<<<256 * 256 / 256, 256, 0, stream>>>(D11, D11T);
  // one-time bf16 copies of the streamed activations
  k_cvt_bf16<<<(NB * NX) / 256, 256, 0, stream>>>(x, xh, NB * NX);
  k_cvt_bf16<<<(NB * NU) / 256, 256, 0, stream>>>(u, uh, NB * NU);

  // xu = x@C1^T + u@D12^T + bv   (grid: 256 m-tiles x N/32 n-tiles, 8 waves/block)
  k_gemm<<<32 * (NQ / 32), 256, 0, stream>>>(xh, C1h, NX, uh, D12h, NU,
                                             nullptr, nullptr, 0, bv, xu, NQ);
  // sequential w solve (writes bf16 w)
  k_solver<<<NB / 32, 256, 0, stream>>>(xu, D11T, lam, wh);
  // dx = x@(Einv F)^T + w@(Einv B1)^T + u@(Einv B2)^T
  k_gemm<<<32 * (NX / 32), 256, 0, stream>>>(xh, G1h, NX, wh, G2h, NQ,
                                             uh, G3h, NU, nullptr, dx, NX);
  // y = x@C2^T + w@D21^T
  k_gemm<<<32 * (NY / 32), 256, 0, stream>>>(xh, C2h, NX, wh, D21h, NQ,
                                             nullptr, nullptr, 0, nullptr, y, NY);
}
